// MLP_42872363549033
// MI455X (gfx1250) — 8-bit matrix operand order repaired and checked by reading, compiled, not run on hardware
//
#include <hip/hip_runtime.h>
#include <hip/hip_bf16.h>

typedef int v8i __attribute__((ext_vector_type(8)));
typedef int i32x4 __attribute__((vector_size(16)));

#define MM 12608
#define DD 768
#define HH 3072

#if __has_builtin(__builtin_amdgcn_global_load_async_to_lds_b128)
#define HAVE_ASYNC_LDS 1
#else
#define HAVE_ASYNC_LDS 0
#endif

__device__ __forceinline__ void copy16_g2l(const signed char* g, signed char* l) {
#if HAVE_ASYNC_LDS
  __builtin_amdgcn_global_load_async_to_lds_b128(
      (__attribute__((address_space(1))) i32x4*)g,
      (__attribute__((address_space(3))) i32x4*)l, 0, 0);
#else
  *(int4*)l = *(const int4*)g;
#endif
}

__device__ __forceinline__ void wait_async0() {
#if HAVE_ASYNC_LDS
#if __has_builtin(__builtin_amdgcn_s_wait_asynccnt)
  __builtin_amdgcn_s_wait_asynccnt(0);
#else
  asm volatile("s_wait_asynccnt 0" ::: "memory");
#endif
#endif
}

__global__ void quant8_kernel(const float* __restrict__ x,
                              const float* __restrict__ Kp,
                              signed char* __restrict__ out, int n) {
  int i = blockIdx.x * blockDim.x + threadIdx.x;
  if (i < n) {
    float K = Kp[0];
    float v = x[i] / K;
    v = fminf(1.0f, fmaxf(-1.0f, v));
    out[i] = (signed char)(int)rintf(v * 127.0f);
  }
}

#define GEMM_CORE(Aptr, Bptr, Kdim)                                            \
  __shared__ __align__(16) signed char Asl[2][64 * 80];                        \
  __shared__ __align__(16) signed char Bsl[2][128 * 80];                       \
  const int tid  = threadIdx.x;                                                \
  const int lane = tid & 31;                                                   \
  const int wid  = tid >> 5;                                                   \
  const int wm   = wid & 3;                                                    \
  const int wn   = wid >> 2;                                                   \
  const int m0   = blockIdx.y * 64;                                            \
  const int n0   = blockIdx.x * 128;                                           \
  const int hi   = (lane >= 16) ? 1 : 0;                                       \
  const int l16  = lane & 15;                                                  \
                             \
  const int arow = tid >> 2, acol = (tid & 3) << 4;                            \
  v8i acc[4] = {};                                                             \
                                   \
  copy16_g2l(&(Aptr)[(size_t)(m0 + arow) * (Kdim) + acol],                     \
             &Asl[0][arow * 80 + acol]);                                       \
  _Pragma("unroll")                                                            \
  for (int r = 0; r < 2; ++r) {                                                \
    int idx = tid + r * 256, brow = idx >> 2, bcol = (idx & 3) << 4;           \
    copy16_g2l(&(Bptr)[(size_t)(n0 + brow) * (Kdim) + bcol],                   \
               &Bsl[0][brow * 80 + bcol]);                                     \
  }                                                                            \
  int buf = 0;                                                                 \
  for (int kk = 0; kk < (Kdim); kk += 64, buf ^= 1) {                          \
    wait_async0();                              \
    __syncthreads();           \
    if (kk + 64 < (Kdim)) {         \
      int nk = kk + 64;                                                        \
      copy16_g2l(&(Aptr)[(size_t)(m0 + arow) * (Kdim) + nk + acol],            \
                 &Asl[buf ^ 1][arow * 80 + acol]);                             \
      _Pragma("unroll")                                                        \
      for (int r = 0; r < 2; ++r) {                                            \
        int idx = tid + r * 256, brow = idx >> 2, bcol = (idx & 3) << 4;       \
        copy16_g2l(&(Bptr)[(size_t)(n0 + brow) * (Kdim) + nk + bcol],          \
                   &Bsl[buf ^ 1][brow * 80 + bcol]);                           \
      }                                                                        \
    }                                                                          \
    const signed char* Ab = &Asl[buf][0];                                      \
    const signed char* Bb = &Bsl[buf][0];                                      \
    v8i af;                                                                    \
    _Pragma("unroll")                                                          \
    for (int j = 0; j < 8; ++j) {                                              \
      int kA = ((j >> 1) << 4) + (hi << 3) + ((j & 1) << 2);                   \
      af[j]  = *(const int*)(Ab + (wm * 16 + l16) * 80 + kA);                  \
    }                                                                          \
    _Pragma("unroll")                                                          \
    for (int t = 0; t < 4; ++t) {                                              \
      v8i bfv;                                                                 \
      _Pragma("unroll")                                                        \
      for (int j = 0; j < 8; ++j) {                                            \
        int kB = ((j >> 1) << 4) + (hi << 3) + ((j & 1) << 2);                 \
        bfv[j] = *(const int*)(Bb + (wn * 64 + t * 16 + l16) * 80 + kB);       \
      }                                                                        \
      acc[t] = __builtin_amdgcn_wmma_i32_16x16x64_iu8(                         \
           true, af,  true, bfv, acc[t],                     \
           false,  false);                               \
    }                                                                          \
  }

__global__ __launch_bounds__(256) void gemm_fc1_kernel(
    const signed char* __restrict__ A,
    const signed char* __restrict__ W,
    const float* __restrict__ b1,
    const float* __restrict__ ka1, const float* __restrict__ kw1,
    const float* __restrict__ ka2,
    signed char* __restrict__ Hq) {
  GEMM_CORE(A, W, DD)
  const float s    = (ka1[0] / 127.0f) * (kw1[0] / 127.0f);
  const float ka2v = ka2[0];
#pragma unroll
  for (int t = 0; t < 4; ++t) {
    const int   col  = n0 + wn * 64 + t * 16 + l16;
    const float bias = b1[col];
#pragma unroll
    for (int r = 0; r < 8; ++r) {
      int   row = m0 + wm * 16 + r + hi * 8;
      float v   = (float)acc[t][r] * s + bias;
      float g   = 0.5f * v * (1.0f + erff(v * 0.70710678118654752f));
      float q   = fminf(1.0f, fmaxf(-1.0f, g / ka2v));
      Hq[(size_t)row * HH + col] = (signed char)(int)rintf(q * 127.0f);
    }
  }
}

__global__ __launch_bounds__(256) void gemm_fc2_kernel(
    const signed char* __restrict__ A,
    const signed char* __restrict__ W,
    const float* __restrict__ b2,
    const float* __restrict__ ka2, const float* __restrict__ kw2,
    float* __restrict__ out) {
  GEMM_CORE(A, W, HH)
  const float s = (ka2[0] / 127.0f) * (kw2[0] / 127.0f);
#pragma unroll
  for (int t = 0; t < 4; ++t) {
    const int   col  = n0 + wn * 64 + t * 16 + l16;
    const float bias = b2[col];
#pragma unroll
    for (int r = 0; r < 8; ++r) {
      int row = m0 + wm * 16 + r + hi * 8;
      out[(size_t)row * DD + col] = (float)acc[t][r] * s + bias;
    }
  }
}

extern "C" void kernel_launch(void* const* d_in, const int* in_sizes, int n_in,
                              void* d_out, int out_size, void* d_ws, size_t ws_size,
                              hipStream_t stream) {
  const float* hs  = (const float*)d_in[0];
  const float* W1  = (const float*)d_in[1];
  const float* b1  = (const float*)d_in[2];
  const float* W2  = (const float*)d_in[3];
  const float* b2  = (const float*)d_in[4];
  const float* ka1 = (const float*)d_in[5];
  const float* kw1 = (const float*)d_in[6];
  const float* ka2 = (const float*)d_in[7];
  const float* kw2 = (const float*)d_in[8];
  float*       out = (float*)d_out;

  signed char* A8  = (signed char*)d_ws;
  signed char* W18 = A8  + (size_t)MM * DD;
  signed char* W28 = W18 + (size_t)HH * DD;
  signed char* H8  = W28 + (size_t)DD * HH;

  quant8_kernel<<<(MM * DD + 255) / 256, 256, 0, stream>>>(hs, ka1, A8, MM * DD);
  quant8_kernel<<<(HH * DD + 255) / 256, 256, 0, stream>>>(W1, kw1, W18, HH * DD);
  quant8_kernel<<<(DD * HH + 255) / 256, 256, 0, stream>>>(W2, kw2, W28, DD * HH);

  gemm_fc1_kernel<<<dim3(HH / 128, MM / 64), 256, 0, stream>>>(
      A8, W18, b1, ka1, kw1, ka2, H8);
  gemm_fc2_kernel<<<dim3(DD / 128, MM / 64), 256, 0, stream>>>(
      H8, W28, b2, ka2, kw2, out);
}
